// DyGKT_5695126634845
// MI455X (gfx1250) — compile-verified, matrix operand order checked by reading, not run on hardware
//
#include <hip/hip_runtime.h>
#include <hip/hip_bf16.h>
#include <stdint.h>

// DyGKT forward for MI455X (gfx1250, wave32, WMMA).
//
// Input indexing assumption: harness flattens the setup_inputs() dict in
// insertion order at top level, with the nested 'params' dict flattened as
// tree leaves in sorted-key order (ASCII; uppercase < lowercase):
//   0 dst_Whh 1 dst_Wih 2 dst_bhh 3 dst_bih 4 edge_W 5 edge_b 6 feat_W
//   7 feat_b 8 node_raw_features 9 out_W 10 out_b 11 src_Whh 12 src_Wih
//   13 src_bhh 14 src_bih 15 struct_W 16 struct_b 17 time_W 18 time_b
//   19 wl_W 20 wl_b 21 wo_W 22 wo_b 23 ws_W 24 ws_b
//   25 src_node_ids 26 dst_node_ids 27 node_interact_times
//   28..30 neighbor_src_{node,edge,time} 31..33 neighbor_dst_{node,edge,time}

typedef __attribute__((ext_vector_type(16))) _Float16 v16h;
typedef __attribute__((ext_vector_type(8)))  _Float16 v8h;
typedef __attribute__((ext_vector_type(8)))  float    v8f;

// ---- workspace layout (in _Float16 elements) ----------------------------
// swizzled f16 B-matrices for WMMA: blob[(kc*numNT+nt)*32 + lane][16 halves]
#define MT_OFF    0       // Mt  : K=128(pad), N=64  -> 4*4*512  = 8192
#define FEAT_OFF  8192    // featW^T: K=64, N=64     -> 2*4*512  = 4096
#define OUT_OFF   12288   // outW^T : K=64, N=64     -> 4096
#define SWIH_OFF  16384   // src Wih^T: K=64, N=192  -> 2*12*512 = 12288
#define SWHH_OFF  28672
#define DWIH_OFF  40960
#define DWHH_OFF  53248
#define WS_HALVES 65536
#define CB_BYTE_OFF 131072  // float cb[64] follows the f16 blobs

__device__ __forceinline__ v8f wmma32(v16h a, v16h b, v8f c) {
  return __builtin_amdgcn_wmma_f32_16x16x32_f16(
      /*neg_a=*/false, a, /*neg_b=*/false, b,
      /*c_mod=*/(short)0, c, /*reuse_a=*/false, /*reuse_b=*/false);
}

// 16x32 f16 A-tile from LDS (row-major, ldr halves/row), per ISA A layout:
// lane<16: row=lane, halves {k0..k0+7, k0+16..k0+23}
// lane>=16: row=lane-16, halves {k0+8..k0+15, k0+24..k0+31}
__device__ __forceinline__ v16h ldsA(const _Float16* base, int ldr, int k0, int lane) {
  int m  = lane & 15;
  int hs = (lane >> 4) << 3;
  const _Float16* p = base + m * ldr + k0 + hs;
  v8h lo = *(const v8h*)p;
  v8h hi = *(const v8h*)(p + 16);
  v16h r;
#pragma unroll
  for (int i = 0; i < 8; ++i) { r[i] = lo[i]; r[i + 8] = hi[i]; }
  return r;
}

// 32x16 f16 B-tile from a pre-swizzled global blob: lane reads 32 contiguous B.
__device__ __forceinline__ v16h gblB(const _Float16* gb, int kc, int nt, int numNT, int lane) {
  const _Float16* p = gb + ((((kc * numNT) + nt) * 32 + lane) << 4);
  v8h lo = *(const v8h*)p;
  v8h hi = *(const v8h*)(p + 8);
  v16h r;
#pragma unroll
  for (int i = 0; i < 8; ++i) { r[i] = lo[i]; r[i + 8] = hi[i]; }
  return r;
}

// ---- prep: fold wo_W into time_W (Mt, cb) and swizzle all weights to f16 --
__global__ void dygkt_prep(const float* __restrict__ feat_W, const float* __restrict__ out_W,
                           const float* __restrict__ sWih, const float* __restrict__ sWhh,
                           const float* __restrict__ dWih, const float* __restrict__ dWhh,
                           const float* __restrict__ time_W, const float* __restrict__ time_b,
                           const float* __restrict__ wo_W, const float* __restrict__ wo_b,
                           _Float16* __restrict__ wsw, float* __restrict__ cb) {
  int e = blockIdx.x * blockDim.x + threadIdx.x;
  if (e < WS_HALVES) {
    int base, numNT, which; const float* W = nullptr;
    if (e < 8192)       { base = MT_OFF;   numNT = 4;  which = 0; }
    else if (e < 12288) { base = FEAT_OFF; numNT = 4;  which = 1; W = feat_W; }
    else if (e < 16384) { base = OUT_OFF;  numNT = 4;  which = 1; W = out_W; }
    else if (e < 28672) { base = SWIH_OFF; numNT = 12; which = 1; W = sWih; }
    else if (e < 40960) { base = SWHH_OFF; numNT = 12; which = 1; W = sWhh; }
    else if (e < 53248) { base = DWIH_OFF; numNT = 12; which = 1; W = dWih; }
    else                { base = DWHH_OFF; numNT = 12; which = 1; W = dWhh; }
    int local  = e - base;
    int tile   = local >> 9;
    int within = local & 511;
    int lane = within >> 4, h = within & 15;
    int kc = tile / numNT, nt = tile % numNT;
    int nl  = lane & 15;
    int hi8 = (lane >> 4) << 3;
    int kl  = (h < 8) ? (hi8 + h) : (16 + hi8 + (h - 8));
    int k = kc * 32 + kl;
    int n = nt * 16 + nl;
    float v = 0.f;
    if (which == 0) {                       // Mt[j][n] = sum_k wo_W[k][j]*time_W[n][k]
      if (k < 100)
        for (int k2 = 0; k2 < 100; ++k2)
          v += wo_W[k2 * 100 + k] * time_W[n * 100 + k2];
    } else {
      v = W[n * 64 + k];                    // B[k][n] = W[n][k]  (W is (N,64))
    }
    wsw[e] = (_Float16)v;
  } else if (e < WS_HALVES + 64) {
    int n = e - WS_HALVES;                  // cb[n] = time_b[n] + wo_b @ time_W[n,:]
    float v = time_b[n];
    for (int k = 0; k < 100; ++k) v += wo_b[k] * time_W[n * 100 + k];
    cb[n] = v;
  }
}

// ---- fused GRU kernel: one block = 16 batch rows of one path ------------
__global__ __launch_bounds__(128)
void dygkt_main(const float* __restrict__ nrf, const float* __restrict__ feat_b,
                const float* __restrict__ edge_W, const float* __restrict__ edge_b,
                const float* __restrict__ struct_W, const float* __restrict__ struct_b,
                const float* __restrict__ out_b,
                const float* __restrict__ sbih, const float* __restrict__ sbhh,
                const float* __restrict__ dbih, const float* __restrict__ dbhh,
                const float* __restrict__ wsW, const float* __restrict__ wsB,
                const float* __restrict__ wlW, const float* __restrict__ wlB,
                const int* __restrict__ src_ids, const int* __restrict__ dst_ids,
                const float* __restrict__ t0,
                const int* __restrict__ ns_node, const float* __restrict__ ns_edge,
                const float* __restrict__ ns_time,
                const int* __restrict__ nd_node, const float* __restrict__ nd_edge,
                const float* __restrict__ nd_time,
                const _Float16* __restrict__ wsw, const float* __restrict__ cbg,
                float* __restrict__ out) {
  __shared__ __align__(16) _Float16 xA[16 * 64];   // GRU input tile (A)
  __shared__ __align__(16) _Float16 hA[16 * 64];   // hidden tile f16 (A)
  __shared__ __align__(16) _Float16 nA[16 * 64];   // gathered node rows (A)
  __shared__ __align__(16) _Float16 sA[16 * 128];  // time-decay relu vec, K-padded
  __shared__ float gx[16 * 192];                   // x @ Wih^T
  __shared__ float gh[16 * 192];                   // h @ Whh^T
  __shared__ float hS[16 * 64];                    // hidden f32
  __shared__ float bihL[192], bhhL[192];
  __shared__ float sWL[64], sBL[64], cbL[64], fbL[64];
  __shared__ float wsWL[100], wsBL[100], wlWL[100], wlBL[100];
  __shared__ float escal[16], coefv[16], diffv[16], cmpf0[16];
  __shared__ int   idsv[16], cmpid[16];

  const int tid  = threadIdx.x;
  const int lane = tid & 31;
  const int wave = tid >> 5;
  const int path = blockIdx.y;          // 0 = src, 1 = dst
  const int b0   = blockIdx.x * 16;

  const int*   nodearr = path ? nd_node : ns_node;
  const float* edgearr = path ? nd_edge : ns_edge;
  const float* timearr = path ? nd_time : ns_time;
  const float* bih = path ? dbih : sbih;
  const float* bhh = path ? dbhh : sbhh;
  const _Float16* wihB = wsw + (path ? DWIH_OFF : SWIH_OFF);
  const _Float16* whhB = wsw + (path ? DWHH_OFF : SWHH_OFF);
  const float sbmul = path ? 1.f : 2.f;  // src: struct(skill)+struct(co) = 2 biases

  for (int i = tid; i < 192; i += 128) { bihL[i] = bih[i]; bhhL[i] = bhh[i]; }
  for (int i = tid; i < 64; i += 128) {
    sWL[i] = struct_W[i]; sBL[i] = struct_b[i]; cbL[i] = cbg[i]; fbL[i] = feat_b[i];
  }
  for (int i = tid; i < 100; i += 128) {
    wsWL[i] = wsW[i]; wsBL[i] = wsB[i]; wlWL[i] = wlW[i]; wlBL[i] = wlB[i];
  }
  for (int i = tid; i < 16 * 64; i += 128) {
    hS[i] = 0.f; hA[i] = (_Float16)0.f; nA[i] = (_Float16)0.f;
  }
  for (int i = tid; i < 16 * 128; i += 128) sA[i] = (_Float16)0.f;
  if (tid < 16) {
    int b = b0 + tid;
    cmpid[tid] = path ? src_ids[b] : dst_ids[b];
    cmpf0[tid] = path ? 0.f : nrf[(size_t)dst_ids[b] * 64];
  }
  __syncthreads();
  const float eW0 = edge_W[0], eB0 = edge_b[0];

  for (int l = 0; l < 200; ++l) {
    // --- per-row scalars ---------------------------------------------------
    if (tid < 16) {
      int b = b0 + tid;
      float tc = timearr[b * 200 + l];
      float tn = (l < 199) ? timearr[b * 200 + l + 1] : t0[b];
      diffv[tid] = tn - tc;
      escal[tid] = edgearr[b * 200 + l] * eW0 + eB0;
      int nid = nodearr[b * 200 + l];
      idsv[tid] = nid;
      float co = (nid == cmpid[tid]) ? 1.f : 0.f;
      if (path == 0) {
        co += (nrf[(size_t)nid * 64] == cmpf0[tid]) ? 1.f : 0.f;   // skill match
        if (l < 199)
          __builtin_prefetch(&nrf[(size_t)nodearr[b * 200 + l + 1] * 64], 0, 0);
      }
      coefv[tid] = co;
    }
    __syncthreads();
    // --- dual-decay relu vector + node-row gather --------------------------
    for (int i = tid; i < 1600; i += 128) {
      int r = i / 100, k = i - r * 100;
      float d = diffv[r];
      float mk = (d > 86400.0f) ? 1.f : 0.f;
      float sh = d * mk * wsWL[k] + wsBL[k];          sh = sh > 0.f ? sh : 0.f;
      float lg = d * (1.f - mk) * wlWL[k] + wlBL[k];  lg = lg > 0.f ? lg : 0.f;
      sA[r * 128 + k] = (_Float16)(sh + lg);
    }
    if (path == 0) {
      for (int i = tid; i < 1024; i += 128) {
        int r = i >> 6, f = i & 63;
        nA[i] = (_Float16)nrf[(size_t)idsv[r] * 64 + f];
      }
    }
    __syncthreads();
    // --- assemble x via WMMA: time proj (+node proj for src) ---------------
    {
      v8f acc = {};
#pragma unroll
      for (int kc = 0; kc < 4; ++kc)
        acc = wmma32(ldsA(sA, 128, kc * 32, lane), gblB(wsw + MT_OFF, kc, wave, 4, lane), acc);
      if (path == 0) {
#pragma unroll
        for (int kc = 0; kc < 2; ++kc)
          acc = wmma32(ldsA(nA, 64, kc * 32, lane), gblB(wsw + FEAT_OFF, kc, wave, 4, lane), acc);
      }
      int n  = wave * 16 + (lane & 15);
      int mb = (lane >> 4) << 3;
      float extn = cbL[n] + sbmul * sBL[n] + (path == 0 ? fbL[n] : 0.f);
#pragma unroll
      for (int i = 0; i < 8; ++i) {
        int m = mb + i;
        xA[m * 64 + n] = (_Float16)(acc[i] + extn + escal[m] + coefv[m] * sWL[n]);
      }
    }
    __syncthreads();
    // --- gate pre-activations: x@Wih^T, h@Whh^T ----------------------------
#pragma unroll
    for (int j = 0; j < 3; ++j) {
      int nt = wave * 3 + j;
      v8f ax = {}, ah = {};
#pragma unroll
      for (int kc = 0; kc < 2; ++kc) {
        ax = wmma32(ldsA(xA, 64, kc * 32, lane), gblB(wihB, kc, nt, 12, lane), ax);
        ah = wmma32(ldsA(hA, 64, kc * 32, lane), gblB(whhB, kc, nt, 12, lane), ah);
      }
      int n  = nt * 16 + (lane & 15);
      int mb = (lane >> 4) << 3;
#pragma unroll
      for (int i = 0; i < 8; ++i) {
        gx[(mb + i) * 192 + n] = ax[i];
        gh[(mb + i) * 192 + n] = ah[i];
      }
    }
    __syncthreads();
    // --- GRU nonlinearity + state update -----------------------------------
    for (int i = tid; i < 1024; i += 128) {
      int r = i >> 6, c = i & 63;
      float xr = gx[r * 192 + c]       + bihL[c];
      float hr = gh[r * 192 + c]       + bhhL[c];
      float xz = gx[r * 192 + 64 + c]  + bihL[64 + c];
      float hz = gh[r * 192 + 64 + c]  + bhhL[64 + c];
      float xn = gx[r * 192 + 128 + c] + bihL[128 + c];
      float hn = gh[r * 192 + 128 + c] + bhhL[128 + c];
      float rg = 1.f / (1.f + __expf(-(xr + hr)));
      float zg = 1.f / (1.f + __expf(-(xz + hz)));
      float ng = tanhf(xn + rg * hn);
      float hv = (1.f - zg) * ng + zg * hS[i];
      hS[i] = hv;
      hA[i] = (_Float16)hv;
    }
    __syncthreads();
  }

  // --- tail: last-slot features (diff=0, edge=0), dst adds node features ---
  if (tid < 16) {
    escal[tid] = eB0;
    idsv[tid]  = dst_ids[b0 + tid];
  }
  __syncthreads();
  for (int i = tid; i < 1600; i += 128) {
    int r = i / 100, k = i - r * 100;
    float sh = wsBL[k] > 0.f ? wsBL[k] : 0.f;
    float lg = wlBL[k] > 0.f ? wlBL[k] : 0.f;
    sA[r * 128 + k] = (_Float16)(sh + lg);
  }
  if (path == 1) {
    for (int i = tid; i < 1024; i += 128) {
      int r = i >> 6, f = i & 63;
      nA[i] = (_Float16)nrf[(size_t)idsv[r] * 64 + f];
    }
  }
  __syncthreads();
  {
    v8f acc = {};
#pragma unroll
    for (int kc = 0; kc < 4; ++kc)
      acc = wmma32(ldsA(sA, 128, kc * 32, lane), gblB(wsw + MT_OFF, kc, wave, 4, lane), acc);
    if (path == 1) {
#pragma unroll
      for (int kc = 0; kc < 2; ++kc)
        acc = wmma32(ldsA(nA, 64, kc * 32, lane), gblB(wsw + FEAT_OFF, kc, wave, 4, lane), acc);
    }
    int n  = wave * 16 + (lane & 15);
    int mb = (lane >> 4) << 3;
    float extn = cbL[n] + (path ? fbL[n] : 0.f);
#pragma unroll
    for (int i = 0; i < 8; ++i) {
      int m = mb + i;
      xA[m * 64 + n] = (_Float16)(hS[m * 64 + n] + acc[i] + extn + escal[m]);
    }
  }
  __syncthreads();
  // --- output projection: emb @ out_W^T + out_b ----------------------------
  {
    v8f acc = {};
#pragma unroll
    for (int kc = 0; kc < 2; ++kc)
      acc = wmma32(ldsA(xA, 64, kc * 32, lane), gblB(wsw + OUT_OFF, kc, wave, 4, lane), acc);
    int n  = wave * 16 + (lane & 15);
    int mb = (lane >> 4) << 3;
#pragma unroll
    for (int i = 0; i < 8; ++i) {
      int m = mb + i;
      out[((size_t)path * 4096 + b0 + m) * 64 + n] = acc[i] + out_b[n];
    }
  }
}

extern "C" void kernel_launch(void* const* d_in, const int* in_sizes, int n_in,
                              void* d_out, int out_size, void* d_ws, size_t ws_size,
                              hipStream_t stream) {
  const float* dst_Whh  = (const float*)d_in[0];
  const float* dst_Wih  = (const float*)d_in[1];
  const float* dst_bhh  = (const float*)d_in[2];
  const float* dst_bih  = (const float*)d_in[3];
  const float* edge_W   = (const float*)d_in[4];
  const float* edge_b   = (const float*)d_in[5];
  const float* feat_W   = (const float*)d_in[6];
  const float* feat_b   = (const float*)d_in[7];
  const float* nrf      = (const float*)d_in[8];
  const float* out_W    = (const float*)d_in[9];
  const float* out_b    = (const float*)d_in[10];
  const float* src_Whh  = (const float*)d_in[11];
  const float* src_Wih  = (const float*)d_in[12];
  const float* src_bhh  = (const float*)d_in[13];
  const float* src_bih  = (const float*)d_in[14];
  const float* struct_W = (const float*)d_in[15];
  const float* struct_b = (const float*)d_in[16];
  const float* time_W   = (const float*)d_in[17];
  const float* time_b   = (const float*)d_in[18];
  const float* wl_W     = (const float*)d_in[19];
  const float* wl_b     = (const float*)d_in[20];
  const float* wo_W     = (const float*)d_in[21];
  const float* wo_b     = (const float*)d_in[22];
  const float* ws_W     = (const float*)d_in[23];
  const float* ws_b     = (const float*)d_in[24];
  const int*   src_ids  = (const int*)d_in[25];
  const int*   dst_ids  = (const int*)d_in[26];
  const float* t0       = (const float*)d_in[27];
  const int*   ns_node  = (const int*)d_in[28];
  const float* ns_edge  = (const float*)d_in[29];
  const float* ns_time  = (const float*)d_in[30];
  const int*   nd_node  = (const int*)d_in[31];
  const float* nd_edge  = (const float*)d_in[32];
  const float* nd_time  = (const float*)d_in[33];
  (void)in_sizes; (void)n_in; (void)out_size; (void)ws_size;

  _Float16* wsw = (_Float16*)d_ws;
  float*    cb  = (float*)((char*)d_ws + CB_BYTE_OFF);

  dygkt_prep<<<(WS_HALVES + 64 + 255) / 256, 256, 0, stream>>>(
      feat_W, out_W, src_Wih, src_Whh, dst_Wih, dst_Whh,
      time_W, time_b, wo_W, wo_b, wsw, cb);

  dim3 grid(4096 / 16, 2, 1);
  dygkt_main<<<grid, 128, 0, stream>>>(
      nrf, feat_b, edge_W, edge_b, struct_W, struct_b, out_b,
      src_bih, src_bhh, dst_bih, dst_bhh, ws_W, ws_b, wl_W, wl_b,
      src_ids, dst_ids, t0, ns_node, ns_edge, ns_time,
      nd_node, nd_edge, nd_time, wsw, cb, (float*)d_out);
}
